// GatedDeltaMemory_10428180594730
// MI455X (gfx1250) — compile-verified, matrix operand order checked by reading, not run on hardware
//
#include <hip/hip_runtime.h>
#include <hip/hip_bf16.h>

#define DEVI __device__ __forceinline__

typedef __attribute__((ext_vector_type(16))) _Float16 v16h;
typedef __attribute__((ext_vector_type(8)))  float    v8f;

// Problem constants (match reference)
constexpr int Bc  = 2;
constexpr int Tc  = 2048;
constexpr int Dc  = 1024;
constexpr int Hc  = 8;
constexpr int DMc = 64;
constexpr int Ec  = Hc * DMc;     // 512
constexpr int Mtot = Bc * Tc;     // 4096

// ---------------------------------------------------------------------------
// WMMA fragment layouts (CDNA5 wave32, ISA 7.12.2)
// A (16x32 f16): lane L holds row M=L&15; hl=L>>4: elems 0..7 = K(8*hl+0..7),
//                elems 8..15 = K(16+8*hl+0..7).
// B (32x16 f16): lane L holds col N=L&15; same K packing as A.
// C (16x16 f32): lane L holds col N=L&15; vgpr j holds row M = j + 8*(L>>4).
//
// Packed-B buffer: fragment f = (k0/32)*(N/16) + (n0/16); lane L's 16 halves
// stored contiguously at ((f*32 + L)*16) -> one 32B vector load per lane.
// ---------------------------------------------------------------------------

DEVI v16h load_a_frag(const _Float16* __restrict__ A, int lda, int m0, int k0, int lane) {
  int m  = m0 + (lane & 15);
  int hl = lane >> 4;
  const _Float16* p0 = A + (size_t)m * lda + (k0 + 8 * hl);
  const _Float16* p1 = p0 + 16;
  v16h r;
#pragma unroll
  for (int j = 0; j < 8; ++j) { r[j] = p0[j]; r[8 + j] = p1[j]; }
  return r;
}

DEVI v16h load_b_packed(const v16h* __restrict__ Bp, int frag, int lane) {
  return Bp[(size_t)frag * 32 + lane];
}

DEVI size_t pack_b_offset(int k, int n, int N) {
  int chunk = k >> 5;
  int kr = k & 31;
  int hl = (kr >> 3) & 1;
  int e  = ((kr >> 4) << 3) | (kr & 7);
  int nt = n >> 4, nl = n & 15;
  return ((size_t)(chunk * (N >> 4) + nt) * 32 + hl * 16 + nl) * 16 + e;
}

DEVI v8f wmma_f16(v16h a, v16h b, v8f c) {
  return __builtin_amdgcn_wmma_f32_16x16x32_f16(false, a, false, b, (short)0, c, false, false);
}

// ---------------------------------------------------------------------------
// Async global->LDS copy of 4KB (8 x b128 per lane), tracked by ASYNCcnt.
// LDS[lds_base + lane*16 + i*512] = MEM[src + lane*16 + i*512]
// ---------------------------------------------------------------------------
DEVI void async_copy_4k(unsigned lds_base, const void* src, int lane) {
  const char* ga = (const char*)src + lane * 16;
  unsigned la = lds_base + lane * 16;
  asm volatile("global_load_async_to_lds_b128 %0, %1, off"              :: "v"(la), "v"(ga) : "memory");
  asm volatile("global_load_async_to_lds_b128 %0, %1, off offset:512"   :: "v"(la), "v"(ga) : "memory");
  asm volatile("global_load_async_to_lds_b128 %0, %1, off offset:1024"  :: "v"(la), "v"(ga) : "memory");
  asm volatile("global_load_async_to_lds_b128 %0, %1, off offset:1536"  :: "v"(la), "v"(ga) : "memory");
  asm volatile("global_load_async_to_lds_b128 %0, %1, off offset:2048"  :: "v"(la), "v"(ga) : "memory");
  asm volatile("global_load_async_to_lds_b128 %0, %1, off offset:2560"  :: "v"(la), "v"(ga) : "memory");
  asm volatile("global_load_async_to_lds_b128 %0, %1, off offset:3072"  :: "v"(la), "v"(ga) : "memory");
  asm volatile("global_load_async_to_lds_b128 %0, %1, off offset:3584"  :: "v"(la), "v"(ga) : "memory");
}
DEVI void wait_async_le8() { asm volatile("s_wait_asynccnt 0x8" ::: "memory"); }
DEVI void wait_async_0()   { asm volatile("s_wait_asynccnt 0x0" ::: "memory"); }

// ---------------------------------------------------------------------------
// Kernel: fp32 -> f16 elementwise (x)
// ---------------------------------------------------------------------------
__global__ void k_cvt_f16(const float* __restrict__ src, _Float16* __restrict__ dst, int n) {
  int i = blockIdx.x * 256 + threadIdx.x;
  if (i < n) dst[i] = (_Float16)src[i];
}

// ---------------------------------------------------------------------------
// Kernel: W [N,K] fp32  ->  packed f16 B buffer for B = W^T [K,N]
// ---------------------------------------------------------------------------
__global__ void k_pack_b_from_wT(const float* __restrict__ W, _Float16* __restrict__ dst,
                                 int K, int N) {
  int i = blockIdx.x * 256 + threadIdx.x;
  if (i >= K * N) return;
  int k = i / N, n = i % N;
  dst[pack_b_offset(k, n, N)] = (_Float16)W[(size_t)n * K + k];
}

// ---------------------------------------------------------------------------
// Kernel: WMMA GEMM  C[M,N] = A[M,K] (f16 row-major) * Bp (packed f16)
// 4 waves / block; each wave computes a 32x64 tile (8 WMMA per 32-K chunk,
// B fragments reused 2x, A fragments 4x).
// ---------------------------------------------------------------------------
__global__ void k_wmma_gemm(const _Float16* __restrict__ A, const _Float16* __restrict__ Bpk,
                            float* __restrict__ C, int M, int N, int K) {
  int lane = threadIdx.x & 31;
  int wave = threadIdx.x >> 5;
  int n0 = blockIdx.x * 64;
  int m0 = blockIdx.y * 128 + wave * 32;
  const v16h* Bp = (const v16h*)Bpk;
  int ntiles = N >> 4;

  v8f acc[2][4] = {};
  for (int k0 = 0; k0 < K; k0 += 32) {
    if (k0 + 32 < K) {
      // prefetch next K-chunk of A rows and packed B (global_prefetch_b8)
      __builtin_prefetch(A + (size_t)(m0 + (lane & 15)) * K + k0 + 32, 0, 1);
      __builtin_prefetch(A + (size_t)(m0 + 16 + (lane & 15)) * K + k0 + 32, 0, 1);
      __builtin_prefetch(&Bp[(size_t)((k0 >> 5) + 1) * ntiles * 32 + (n0 >> 4) * 32 + lane], 0, 1);
    }
    v16h a0 = load_a_frag(A, K, m0,      k0, lane);
    v16h a1 = load_a_frag(A, K, m0 + 16, k0, lane);
    int fb  = (k0 >> 5) * ntiles + (n0 >> 4);
#pragma unroll
    for (int nt = 0; nt < 4; ++nt) {
      v16h b = load_b_packed(Bp, fb + nt, lane);
      acc[0][nt] = wmma_f16(a0, b, acc[0][nt]);
      acc[1][nt] = wmma_f16(a1, b, acc[1][nt]);
    }
  }
  int hl = lane >> 4, nl = lane & 15;
#pragma unroll
  for (int mt = 0; mt < 2; ++mt)
#pragma unroll
    for (int nt = 0; nt < 4; ++nt)
#pragma unroll
      for (int j = 0; j < 8; ++j) {
        int m = m0 + mt * 16 + j + 8 * hl;
        C[(size_t)m * N + n0 + nt * 16 + nl] = acc[mt][nt][j];
      }
}

// ---------------------------------------------------------------------------
// Kernel: beta = sigmoid(x @ Wb^T) and cum = cumsum(log(1-beta+1e-8)) per (b,h)
// ---------------------------------------------------------------------------
__global__ void k_beta_cumsum(const float* __restrict__ x, const float* __restrict__ Wb,
                              float* __restrict__ beta, float* __restrict__ cum) {
  int bh = blockIdx.x;           // b*H + h
  int b = bh / Hc, h = bh % Hc;
  int tid = threadIdx.x;
  __shared__ float slr[Tc];
  __shared__ float spart[256];
  __shared__ float soffs[256];

  const float* wb = Wb + (size_t)h * Dc;
  for (int t = tid; t < Tc; t += 256) {
    const float* xr = x + ((size_t)b * Tc + t) * Dc;
    float acc = 0.f;
    for (int d = 0; d < Dc; ++d) acc = fmaf(xr[d], wb[d], acc);
    float bv = 1.0f / (1.0f + __expf(-acc));
    beta[(size_t)bh * Tc + t] = bv;
    slr[t] = __logf(1.0f - bv + 1e-8f);
  }
  __syncthreads();

  float s = 0.f;
  int base = tid * 8;
#pragma unroll
  for (int i = 0; i < 8; ++i) s += slr[base + i];
  spart[tid] = s;
  __syncthreads();
  if (tid == 0) {
    float run = 0.f;
    for (int i = 0; i < 256; ++i) { soffs[i] = run; run += spart[i]; }
  }
  __syncthreads();
  float run = soffs[tid];
#pragma unroll
  for (int i = 0; i < 8; ++i) {
    run += slr[base + i];
    cum[(size_t)bh * Tc + base + i] = run;
  }
}

// ---------------------------------------------------------------------------
// Kernel: L2-normalize Q (fold 1/sqrt(DM)) and K, pack f16.
// qh    : [B,H,T,DM] row-major          (A-matrix for Q K^T)
// kpack : per (b,h) packed-B, B=K^T (K-dim=DM, N-dim=T)
// vpack : per (b,h) packed-B, B=V   (K-dim=T,  N-dim=DM)
// ---------------------------------------------------------------------------
__global__ void k_normalize_pack(const float* __restrict__ Yq, const float* __restrict__ Yk,
                                 const float* __restrict__ Yv,
                                 _Float16* __restrict__ qh, _Float16* __restrict__ kpack,
                                 _Float16* __restrict__ vpack) {
  int idx = blockIdx.x * 256 + threadIdx.x;    // over B*H*T
  if (idx >= Bc * Hc * Tc) return;
  int t = idx % Tc;
  int bh = idx / Tc;
  int h = bh % Hc, b = bh / Hc;
  size_t row = (size_t)b * Tc + t;
  const float* yq = Yq + row * Ec + h * DMc;
  const float* yk = Yk + row * Ec + h * DMc;
  const float* yv = Yv + row * Ec + h * DMc;

  float nq = 0.f, nk = 0.f;
#pragma unroll 4
  for (int d = 0; d < DMc; ++d) { nq = fmaf(yq[d], yq[d], nq); nk = fmaf(yk[d], yk[d], nk); }
  float sq = 0.125f / fmaxf(__fsqrt_rn(nq), 1e-12f);   // 1/sqrt(DM) folded into Q
  float sk = 1.0f   / fmaxf(__fsqrt_rn(nk), 1e-12f);

  _Float16* qrow = qh + ((size_t)bh * Tc + t) * DMc;
  _Float16* kp   = kpack + (size_t)bh * DMc * Tc;
  _Float16* vp   = vpack + (size_t)bh * Tc * DMc;
#pragma unroll 4
  for (int d = 0; d < DMc; ++d) {
    qrow[d] = (_Float16)(yq[d] * sq);
    kp[pack_b_offset(d, t, Tc)]  = (_Float16)(yk[d] * sk);   // B element (k=d, n=t)
    vp[pack_b_offset(t, d, DMc)] = (_Float16)yv[d];          // B element (k=t, n=d)
  }
}

// ---------------------------------------------------------------------------
// Kernel: fused decay attention.
// Per (b,h) and 16-query tile: stream 32-wide key chunks s >= t-tile,
//   S = (Q K^T) .* exp(min(cum_t - cum_s,0)) .* [s>=t]  (1/sqrt(DM) in Q)
//   O += S @ V ; epilogue O += beta*V (fp32 V) ; store f16 to attn_out [B*T,E]
// one wave per block. V fragments are double-buffered in LDS via
// global_load_async_to_lds_b128 (ASYNCcnt), overlapping with score compute.
// ---------------------------------------------------------------------------
__global__ void k_attention(const _Float16* __restrict__ qh, const _Float16* __restrict__ kpack,
                            const _Float16* __restrict__ vpack, const float* __restrict__ Yv,
                            const float* __restrict__ beta, const float* __restrict__ cum,
                            _Float16* __restrict__ attn_out) {
  int lane = threadIdx.x & 31;
  int t0 = blockIdx.x * 16;
  int bh = blockIdx.y;
  int b = bh / Hc, h = bh % Hc;

  const _Float16* Q  = qh + (size_t)bh * Tc * DMc;                 // [T, DM]
  const v16h* Kp = (const v16h*)(kpack + (size_t)bh * DMc * Tc);   // packed, N-dim = T
  const v16h* Vp = (const v16h*)(vpack + (size_t)bh * Tc * DMc);   // packed, N-dim = DM
  const float* cumv  = cum  + (size_t)bh * Tc;
  const float* betav = beta + (size_t)bh * Tc;

  __shared__ float sS[16 * 32];          // score repack buffer (one wave/block)
  __shared__ _Float16 sV[2][2048];       // double-buffered V chunk (2 x 4KB)

  int hl = lane >> 4, nl = lane & 15;
  const int kNT = Tc >> 4;    // 128 n-tiles in packed K
  const int vNT = DMc >> 4;   // 4  n-tiles in packed V

  unsigned lds_v = (unsigned)(uintptr_t)&sV[0][0];

  // Q A-fragments for the two 32-wide dm chunks (reused across all key tiles)
  v16h aq0 = load_a_frag(Q, DMc, t0, 0,  lane);
  v16h aq1 = load_a_frag(Q, DMc, t0, 32, lane);

  // cum for the 8 query rows this lane covers in the C layout
  float cum_q[8];
#pragma unroll
  for (int j = 0; j < 8; ++j) cum_q[j] = cumv[t0 + j + 8 * hl];

  v8f acc0 = {}, acc1 = {}, acc2 = {}, acc3 = {};

  int s_start = t0 & ~31;                // first chunk touching s >= t0
  int buf = 0;
  // preload first V chunk (4KB contiguous in packed buffer)
  async_copy_4k(lds_v, Vp + (size_t)(s_start >> 5) * vNT * 32, lane);

  for (int s0 = s_start; s0 < Tc; s0 += 32) {
    bool have_next = (s0 + 32 < Tc);
    if (have_next)
      async_copy_4k(lds_v + (buf ^ 1) * 4096, Vp + (size_t)((s0 + 32) >> 5) * vNT * 32, lane);

    // two 16x16 score tiles -> decayed fp32 chunk in LDS (m, s_local) layout
#pragma unroll
    for (int st = 0; st < 2; ++st) {
      int sc = s0 + st * 16;
      v16h kb0 = load_b_packed(Kp, 0 * kNT + (sc >> 4), lane);   // dm chunk 0
      v16h kb1 = load_b_packed(Kp, 1 * kNT + (sc >> 4), lane);   // dm chunk 1
      v8f sacc = {};
      sacc = wmma_f16(aq0, kb0, sacc);
      sacc = wmma_f16(aq1, kb1, sacc);
      int s = sc + nl;
      float cs = cumv[s];
#pragma unroll
      for (int j = 0; j < 8; ++j) {
        int tq = t0 + j + 8 * hl;
        float dec = (s >= tq) ? __expf(fminf(cum_q[j] - cs, 0.f)) : 0.f;
        sS[(j + 8 * hl) * 32 + st * 16 + nl] = sacc[j] * dec;
      }
    }
    __builtin_amdgcn_wave_barrier();
    // repack C-layout scores -> A-layout f16 fragment (cross-lane via LDS)
    v16h as;
#pragma unroll
    for (int j = 0; j < 8; ++j) {
      as[j]     = (_Float16)sS[nl * 32 + 8 * hl + j];
      as[8 + j] = (_Float16)sS[nl * 32 + 16 + 8 * hl + j];
    }
    __builtin_amdgcn_wave_barrier();

    // wait for this chunk's async V copy (in-order completion => <=8 outstanding
    // means the batch before the just-issued one has landed)
    if (have_next) wait_async_le8(); else wait_async_0();

    // S @ V from LDS (ds_load_b128 pairs), 4 dm tiles
    const v16h* lv = (const v16h*)&sV[buf][0];
    v16h bv0 = lv[0 * 32 + lane];
    v16h bv1 = lv[1 * 32 + lane];
    v16h bv2 = lv[2 * 32 + lane];
    v16h bv3 = lv[3 * 32 + lane];
    acc0 = wmma_f16(as, bv0, acc0);
    acc1 = wmma_f16(as, bv1, acc1);
    acc2 = wmma_f16(as, bv2, acc2);
    acc3 = wmma_f16(as, bv3, acc3);
    buf ^= 1;
  }

  // epilogue: O += beta * V (fp32 V from projection), store f16 to attn_out [B*T, E]
  v8f accs[4] = {acc0, acc1, acc2, acc3};
#pragma unroll
  for (int nt = 0; nt < 4; ++nt) {
#pragma unroll
    for (int j = 0; j < 8; ++j) {
      int m  = t0 + j + 8 * hl;
      int dm = nt * 16 + nl;
      size_t row = (size_t)b * Tc + m;
      float o = accs[nt][j] + betav[m] * Yv[row * Ec + h * DMc + dm];
      attn_out[row * Ec + h * DMc + dm] = (_Float16)o;
    }
  }
}

// ---------------------------------------------------------------------------
// Host launcher
// ---------------------------------------------------------------------------
extern "C" void kernel_launch(void* const* d_in, const int* in_sizes, int n_in,
                              void* d_out, int out_size, void* d_ws, size_t ws_size,
                              hipStream_t stream) {
  const float* x  = (const float*)d_in[0];   // [B,T,D]
  const float* Wk = (const float*)d_in[1];   // [E,D]
  const float* Wv = (const float*)d_in[2];   // [E,D]
  const float* Wq = (const float*)d_in[3];   // [E,D]
  const float* Wb = (const float*)d_in[4];   // [H,D]
  const float* Wo = (const float*)d_in[5];   // [D,E]
  float* out = (float*)d_out;                // [B,T,D]

  // workspace carve-up
  char* w = (char*)d_ws;
  auto alloc = [&](size_t bytes) -> void* {
    void* p = (void*)w;
    w += (bytes + 255) & ~(size_t)255;
    return p;
  };
  _Float16* xh   = (_Float16*)alloc((size_t)Mtot * Dc * 2);   // x as f16 row-major
  _Float16* wkP  = (_Float16*)alloc((size_t)Dc * Ec * 2);     // packed B = Wk^T
  _Float16* wvP  = (_Float16*)alloc((size_t)Dc * Ec * 2);
  _Float16* wqP  = (_Float16*)alloc((size_t)Dc * Ec * 2);
  _Float16* woP  = (_Float16*)alloc((size_t)Ec * Dc * 2);     // packed B = Wo^T
  float*    Yq   = (float*)alloc((size_t)Mtot * Ec * 4);
  float*    Yk   = (float*)alloc((size_t)Mtot * Ec * 4);
  float*    Yv   = (float*)alloc((size_t)Mtot * Ec * 4);
  _Float16* qh   = (_Float16*)alloc((size_t)Bc * Hc * Tc * DMc * 2);
  _Float16* kpack= (_Float16*)alloc((size_t)Bc * Hc * Tc * DMc * 2);
  _Float16* vpack= (_Float16*)alloc((size_t)Bc * Hc * Tc * DMc * 2);
  float*    beta = (float*)alloc((size_t)Bc * Hc * Tc * 4);
  float*    cum  = (float*)alloc((size_t)Bc * Hc * Tc * 4);
  _Float16* aoh  = (_Float16*)alloc((size_t)Mtot * Ec * 2);   // attention out f16

  // 1) convert x; transpose+pack weights into fragment-major B buffers
  {
    int n = Mtot * Dc;
    k_cvt_f16<<<dim3((n + 255) / 256), dim3(256), 0, stream>>>(x, xh, n);
    int nw = Ec * Dc;
    dim3 g((nw + 255) / 256);
    k_pack_b_from_wT<<<g, dim3(256), 0, stream>>>(Wk, wkP, Dc, Ec);  // B=[D,E]
    k_pack_b_from_wT<<<g, dim3(256), 0, stream>>>(Wv, wvP, Dc, Ec);
    k_pack_b_from_wT<<<g, dim3(256), 0, stream>>>(Wq, wqP, Dc, Ec);
    k_pack_b_from_wT<<<g, dim3(256), 0, stream>>>(Wo, woP, Ec, Dc);  // B=[E,D]
  }

  // 2) projections: Y = xh @ W^T  (M=4096, N=512, K=1024)
  {
    dim3 grid(Ec / 64, Mtot / 128);
    k_wmma_gemm<<<grid, dim3(128), 0, stream>>>(xh, wqP, Yq, Mtot, Ec, Dc);
    k_wmma_gemm<<<grid, dim3(128), 0, stream>>>(xh, wkP, Yk, Mtot, Ec, Dc);
    k_wmma_gemm<<<grid, dim3(128), 0, stream>>>(xh, wvP, Yv, Mtot, Ec, Dc);
  }

  // 3) beta + cumsum per (b,h)
  k_beta_cumsum<<<dim3(Bc * Hc), dim3(256), 0, stream>>>(x, Wb, beta, cum);

  // 4) normalize + pack f16 (Q row-major+scaled, K/V fragment-packed)
  k_normalize_pack<<<dim3((Bc * Hc * Tc) / 256), dim3(256), 0, stream>>>(Yq, Yk, Yv, qh, kpack, vpack);

  // 5) fused decay attention (async double-buffered V via ASYNCcnt)
  k_attention<<<dim3(Tc / 16, Bc * Hc), dim3(32), 0, stream>>>(qh, kpack, vpack, Yv, beta, cum, aoh);

  // 6) output projection: out = aoh @ Wo^T  (M=4096, N=1024, K=512)
  {
    dim3 grid(Dc / 64, Mtot / 128);
    k_wmma_gemm<<<grid, dim3(128), 0, stream>>>(aoh, woP, out, Mtot, Dc, Ec);
  }
}
